// TransformerBlock_21431886807770
// MI455X (gfx1250) — compile-verified, matrix operand order checked by reading, not run on hardware
//
#include <hip/hip_runtime.h>

typedef __attribute__((ext_vector_type(16))) __bf16 v16bf;
typedef __attribute__((ext_vector_type(8)))  __bf16 v8bf;
typedef __attribute__((ext_vector_type(8)))  float  v8f;

// ---------------------------------------------------------------------------
// Fragment loader: 16x32 A-layout (or B-from-row-major-contraction layout).
// lane 0-15 : row = row0+lane, K = [k0+0..7]   (elems 0-7),  K=[k0+16..23] (8-15)
// lane16-31 : row = row0+lane-16, K=[k0+8..15] (elems 0-7),  K=[k0+24..31] (8-15)
// -> two 16B loads per lane (ds_load_b128 / global_load_b128 after inlining).
// ---------------------------------------------------------------------------
__device__ __forceinline__ v16bf frag16(const __bf16* base, int row0, int stride,
                                        int k0, int lane) {
  const __bf16* p = base + (size_t)(row0 + (lane & 15)) * stride + k0 + ((lane >> 4) << 3);
  v8bf lo = *(const v8bf*)p;
  v8bf hh = *(const v8bf*)(p + 16);
  v16bf f;
#pragma unroll
  for (int i = 0; i < 8; ++i) { f[i] = lo[i]; f[i + 8] = hh[i]; }
  return f;
}

__device__ __forceinline__ v8f wmma_bf16(v16bf a, v16bf b, v8f c) {
  return __builtin_amdgcn_wmma_f32_16x16x32_bf16(false, a, false, b, (short)0, c,
                                                 false, false);
}

__device__ __forceinline__ v8f v8f_zero() {
  v8f z = {0.f, 0.f, 0.f, 0.f, 0.f, 0.f, 0.f, 0.f};
  return z;
}

// ---------------------------------------------------------------------------
// CDNA5 async global->LDS copy (ASYNCcnt-tracked), bypassing VGPRs.
// Generic pointers to __shared__ carry the LDS byte offset in their low 32
// bits (LDS aperture), so the cast yields the required LDS address operand.
// ---------------------------------------------------------------------------
__device__ __forceinline__ void async_ld_b128(void* lds_ptr, const void* gptr) {
  uint32_t lds_addr = (uint32_t)(uintptr_t)lds_ptr;
  uint64_t gaddr = (uint64_t)(uintptr_t)gptr;
  asm volatile("global_load_async_to_lds_b128 %0, %1, off"
               :: "v"(lds_addr), "v"(gaddr)
               : "memory");
}

__device__ __forceinline__ void wait_async0() {
  asm volatile("s_wait_asynccnt 0" ::: "memory");
}

// ---------------------------------------------------------------------------
// fp32 -> bf16 cast (weights)
// ---------------------------------------------------------------------------
__global__ void tb_cast_bf16(const float* __restrict__ in, __bf16* __restrict__ out,
                             int n) {
  int i = blockIdx.x * blockDim.x + threadIdx.x;
  if (i < n) out[i] = (__bf16)in[i];
}

// ---------------------------------------------------------------------------
// LayerNorm over rows of 512 fp32 -> bf16.  One 128-thread block per row.
// ---------------------------------------------------------------------------
__global__ void __launch_bounds__(128)
tb_layernorm(const float* __restrict__ X, const float* __restrict__ w,
             const float* __restrict__ bb, __bf16* __restrict__ out) {
  const int row = blockIdx.x, tid = threadIdx.x;
  const float4 v = ((const float4*)(X + (size_t)row * 512))[tid];
  float s  = v.x + v.y + v.z + v.w;
  float ss = v.x * v.x + v.y * v.y + v.z * v.z + v.w * v.w;
#pragma unroll
  for (int off = 1; off < 32; off <<= 1) {
    s  += __shfl_xor(s,  off, 32);
    ss += __shfl_xor(ss, off, 32);
  }
  __shared__ float sb[4], ssb[4];
  if ((tid & 31) == 0) { sb[tid >> 5] = s; ssb[tid >> 5] = ss; }
  __syncthreads();
  s  = sb[0] + sb[1] + sb[2] + sb[3];
  ss = ssb[0] + ssb[1] + ssb[2] + ssb[3];
  const float mu   = s * (1.0f / 512.0f);
  const float var  = ss * (1.0f / 512.0f) - mu * mu;
  const float rstd = rsqrtf(var + 1e-5f);
  const float4 wv = ((const float4*)w)[tid];
  const float4 bv = ((const float4*)bb)[tid];
  __bf16* op = out + (size_t)row * 512 + tid * 4;
  op[0] = (__bf16)((v.x - mu) * rstd * wv.x + bv.x);
  op[1] = (__bf16)((v.y - mu) * rstd * wv.y + bv.y);
  op[2] = (__bf16)((v.z - mu) * rstd * wv.z + bv.z);
  op[3] = (__bf16)((v.w - mu) * rstd * wv.w + bv.w);
}

// ---------------------------------------------------------------------------
// In-place RoPE on bf16 [B*S, 512].  One thread per (row, d<256) pair.
// ---------------------------------------------------------------------------
__global__ void tb_rope(__bf16* __restrict__ X, int S, int total) {
  int idx = blockIdx.x * blockDim.x + threadIdx.x;
  if (idx >= total) return;
  const int d  = idx & 255;
  const int rs = idx >> 8;           // flattened row (b*S + s)
  const int s  = rs % S;
  const float inv = __expf(-(float)d * (9.210340371976184f / 256.0f)); // 10000^(-d/256)
  const float ang = (float)s * inv;
  float sn, cs;
  __sincosf(ang, &sn, &cs);
  __bf16* p = X + (size_t)rs * 512;
  const float x1 = (float)p[d];
  const float x2 = (float)p[d + 256];
  p[d]       = (__bf16)(x1 * cs - x2 * sn);
  p[d + 256] = (__bf16)(x2 * cs + x1 * sn);
}

// ---------------------------------------------------------------------------
// WMMA GEMM:  C[M,N] = A[M,K](bf16) * B[K,N](bf16) + bias, with epilogue:
//   MODE 0: store bf16           MODE 1: ReLU, store bf16
//   MODE 2: += R[M,N] (fp32), store fp32
// BM=64 BN=128 BK=64, 256 threads = 8 waves, each wave a 32x32 C tile.
// A tile staged via async global->LDS; B tile transposed manually.
// ---------------------------------------------------------------------------
template <int MODE>
__global__ void __launch_bounds__(256)
tb_gemm(const __bf16* __restrict__ A, const __bf16* __restrict__ B,
        const float* __restrict__ bias, const float* __restrict__ R,
        void* __restrict__ Cout, int M, int N, int K) {
  constexpr int BM = 64, BN = 128, BK = 64, LDA = BK + 8, LDB = BK + 8;
  __shared__ __bf16 As[BM * LDA];    // 9216 B
  __shared__ __bf16 Bt[BN * LDB];    // 18432 B (B tile stored transposed [n][k])
  const int tid = threadIdx.x, lane = tid & 31, wid = tid >> 5;
  const int bm0 = blockIdx.y * BM, bn0 = blockIdx.x * BN;
  const int wm = (wid >> 2) * 32, wn = (wid & 3) * 32;

  v8f acc[2][2];
#pragma unroll
  for (int i = 0; i < 2; ++i)
#pragma unroll
    for (int j = 0; j < 2; ++j) acc[i][j] = v8f_zero();

  for (int k0 = 0; k0 < K; k0 += BK) {
    __syncthreads();
    // A tile: 64x64 halves, contiguous 16B chunks, async direct-to-LDS.
#pragma unroll
    for (int i = 0; i < 2; ++i) {
      int c = tid + 256 * i;               // 0..511
      int row = c >> 3, col = (c & 7) << 3;
      async_ld_b128(As + row * LDA + col,
                    A + (size_t)(bm0 + row) * K + k0 + col);
    }
    // B tile: 64x128 halves, read contiguous along N, scatter transposed.
#pragma unroll
    for (int i = 0; i < 4; ++i) {
      int c = tid + 256 * i;               // 0..1023
      int kk = c >> 4, n8 = (c & 15) << 3;
      v8bf t = *(const v8bf*)(B + (size_t)(k0 + kk) * N + bn0 + n8);
#pragma unroll
      for (int j = 0; j < 8; ++j) Bt[(n8 + j) * LDB + kk] = t[j];
    }
    // Prefetch next K panel toward cache while this tile computes.
    if (k0 + BK < K) {
      __builtin_prefetch(A + (size_t)(bm0 + (tid >> 2)) * K + k0 + BK, 0, 3);
      __builtin_prefetch(B + (size_t)(k0 + BK + (tid >> 2)) * N + bn0, 0, 3);
    }
    wait_async0();
    __syncthreads();
#pragma unroll
    for (int ks = 0; ks < BK; ks += 32) {
      v16bf a0 = frag16(As, wm,      LDA, ks, lane);
      v16bf a1 = frag16(As, wm + 16, LDA, ks, lane);
      v16bf b0 = frag16(Bt, wn,      LDB, ks, lane);
      v16bf b1 = frag16(Bt, wn + 16, LDB, ks, lane);
      acc[0][0] = wmma_bf16(a0, b0, acc[0][0]);
      acc[0][1] = wmma_bf16(a0, b1, acc[0][1]);
      acc[1][0] = wmma_bf16(a1, b0, acc[1][0]);
      acc[1][1] = wmma_bf16(a1, b1, acc[1][1]);
    }
  }

  const int hi = lane >> 4, ln16 = lane & 15;
#pragma unroll
  for (int i = 0; i < 2; ++i)
#pragma unroll
    for (int j = 0; j < 2; ++j) {
      const int mb = bm0 + wm + 16 * i + hi * 8;
      const int n  = bn0 + wn + 16 * j + ln16;
      const float bvv = bias[n];
#pragma unroll
      for (int r = 0; r < 8; ++r) {
        const int m = mb + r;
        float v = acc[i][j][r] + bvv;
        if (MODE == 1) v = v > 0.f ? v : 0.f;
        if (MODE == 2) {
          ((float*)Cout)[(size_t)m * N + n] = v + R[(size_t)m * N + n];
        } else {
          ((__bf16*)Cout)[(size_t)m * N + n] = (__bf16)v;
        }
      }
    }
}

// ---------------------------------------------------------------------------
// Flash attention, D=512, KV tile = 32 rows.  128 threads = 4 waves; each
// wave owns 16 q rows: Q frags (128 VGPRs) + 16x512 fp32 O accum (256 VGPRs).
// K tile staged via async global->LDS; V staged manually (needs transpose).
// P is transposed C-layout -> A-layout through a per-wave LDS scratch that
// aliases the K tile (barrier-protected).
// ---------------------------------------------------------------------------
__global__ void __launch_bounds__(128, 1)
tb_flash_attn(const __bf16* __restrict__ Qg, const __bf16* __restrict__ Kg,
              const __bf16* __restrict__ Vg, __bf16* __restrict__ Og, int S) {
  __shared__ __bf16 Ks[32 * 512];   // 32 KB  (K tile, row-major [kv][d])
  __shared__ __bf16 Vt[512 * 32];   // 32 KB  (V tile, transposed [d][kv])
  const int tid = threadIdx.x, lane = tid & 31, wid = tid >> 5;
  const int hi = lane >> 4, ln16 = lane & 15;
  const int b = blockIdx.y;
  const int q0 = blockIdx.x * 64 + wid * 16;
  const __bf16* Qb = Qg + (size_t)b * S * 512;
  const __bf16* Kb = Kg + (size_t)b * S * 512;
  const __bf16* Vb = Vg + (size_t)b * S * 512;

  v16bf qf[16];
#pragma unroll
  for (int s = 0; s < 16; ++s) qf[s] = frag16(Qb, q0, 512, 32 * s, lane);

  v8f o[32];
#pragma unroll
  for (int j = 0; j < 32; ++j) o[j] = v8f_zero();
  float mr[8], lr[8];
#pragma unroll
  for (int r = 0; r < 8; ++r) { mr[r] = -3.0e38f; lr[r] = 0.f; }

  const float sc = 0.044194173824159216f;  // 1/sqrt(512)
  __bf16* Ps = Ks + wid * 512;             // per-wave 16x32 scratch (aliases Ks)

  for (int kv0 = 0; kv0 < S; kv0 += 32) {
    __syncthreads();  // previous iteration's Vt/Ps reads complete
    // Stage K (async direct-to-LDS) and V (transposed, manual): 2048 chunks.
#pragma unroll
    for (int i = 0; i < 16; ++i) {
      int c = tid + 128 * i;
      int row = c >> 6, c8 = (c & 63) << 3;
      async_ld_b128(Ks + row * 512 + c8,
                    Kb + (size_t)(kv0 + row) * 512 + c8);
      v8bf tv = *(const v8bf*)(Vb + (size_t)(kv0 + row) * 512 + c8);
#pragma unroll
      for (int jj = 0; jj < 8; ++jj) Vt[(c8 + jj) * 32 + row] = tv[jj];
    }
    wait_async0();
    __syncthreads();

    // S = Q * K^T for this wave's 16 q rows x 32 kv cols.
    v8f s0 = v8f_zero(), s1 = v8f_zero();
#pragma unroll
    for (int s = 0; s < 16; ++s) {
      v16bf kb0 = frag16(Ks, 0,  512, 32 * s, lane);
      v16bf kb1 = frag16(Ks, 16, 512, 32 * s, lane);
      s0 = wmma_bf16(qf[s], kb0, s0);
      s1 = wmma_bf16(qf[s], kb1, s1);
    }
    __syncthreads();  // all waves done reading Ks before Ps (alias) is written

    // Online softmax over the 32 new columns (row reduce across 16-lane half).
#pragma unroll
    for (int r = 0; r < 8; ++r) {
      float v0 = s0[r] * sc, v1 = s1[r] * sc;
      float mx = fmaxf(v0, v1);
#pragma unroll
      for (int off = 1; off < 16; off <<= 1) mx = fmaxf(mx, __shfl_xor(mx, off, 32));
      const float mnew  = fmaxf(mr[r], mx);
      const float alpha = __expf(mr[r] - mnew);
      mr[r] = mnew;
      const float p0 = __expf(v0 - mnew), p1 = __expf(v1 - mnew);
      float rs = p0 + p1;
#pragma unroll
      for (int off = 1; off < 16; off <<= 1) rs += __shfl_xor(rs, off, 32);
      lr[r] = lr[r] * alpha + rs;
      const int row = r + 8 * hi;
      Ps[row * 32 + ln16]      = (__bf16)p0;
      Ps[row * 32 + 16 + ln16] = (__bf16)p1;
#pragma unroll
      for (int j = 0; j < 32; ++j) o[j][r] *= alpha;
    }
    asm volatile("s_wait_dscnt 0" ::: "memory");  // P stores visible to own wave
    v16bf pf = frag16(Ps, 0, 32, 0, lane);
    // O += P * V over all 32 d-tiles.
#pragma unroll
    for (int j = 0; j < 32; ++j) {
      v16bf vb = frag16(Vt, 16 * j, 32, 0, lane);
      o[j] = wmma_bf16(pf, vb, o[j]);
    }
  }

  // Normalize and store.
#pragma unroll
  for (int j = 0; j < 32; ++j)
#pragma unroll
    for (int r = 0; r < 8; ++r) {
      const int row = q0 + r + 8 * hi;
      Og[((size_t)b * S + row) * 512 + 16 * j + ln16] = (__bf16)(o[j][r] / lr[r]);
    }
}

// ---------------------------------------------------------------------------
// Host launcher
// ---------------------------------------------------------------------------
extern "C" void kernel_launch(void* const* d_in, const int* in_sizes, int n_in,
                              void* d_out, int out_size, void* d_ws, size_t ws_size,
                              hipStream_t stream) {
  const float* q    = (const float*)d_in[0];
  const float* k    = (const float*)d_in[1];
  const float* Wq   = (const float*)d_in[2];
  const float* bq   = (const float*)d_in[3];
  const float* Wk   = (const float*)d_in[4];
  const float* bk   = (const float*)d_in[5];
  const float* Wv   = (const float*)d_in[6];
  const float* bv   = (const float*)d_in[7];
  const float* Wo   = (const float*)d_in[8];
  const float* bo   = (const float*)d_in[9];
  const float* lnqw = (const float*)d_in[10];
  const float* lnqb = (const float*)d_in[11];
  const float* lnkw = (const float*)d_in[12];
  const float* lnkb = (const float*)d_in[13];
  const float* lnow = (const float*)d_in[14];
  const float* lnob = (const float*)d_in[15];
  const float* W1   = (const float*)d_in[16];
  const float* b1   = (const float*)d_in[17];
  const float* W2   = (const float*)d_in[18];
  const float* b2   = (const float*)d_in[19];

  constexpr int B = 2, S = 4096, Dm = 512, DF = 2048;
  constexpr int M = B * S;  // 8192

  char* w = (char*)d_ws;
  auto alloc = [&](size_t bytes) -> char* {
    char* p = w;
    w += (bytes + 255) & ~(size_t)255;
    return p;
  };
  __bf16* Wq_b = (__bf16*)alloc((size_t)Dm * Dm * 2);
  __bf16* Wk_b = (__bf16*)alloc((size_t)Dm * Dm * 2);
  __bf16* Wv_b = (__bf16*)alloc((size_t)Dm * Dm * 2);
  __bf16* Wo_b = (__bf16*)alloc((size_t)Dm * Dm * 2);
  __bf16* W1_b = (__bf16*)alloc((size_t)Dm * DF * 2);
  __bf16* W2_b = (__bf16*)alloc((size_t)DF * Dm * 2);
  __bf16* qn   = (__bf16*)alloc((size_t)M * Dm * 2);
  __bf16* kn   = (__bf16*)alloc((size_t)M * Dm * 2);
  __bf16* Qp   = (__bf16*)alloc((size_t)M * Dm * 2);
  __bf16* Kp   = (__bf16*)alloc((size_t)M * Dm * 2);
  __bf16* Vp   = (__bf16*)alloc((size_t)M * Dm * 2);
  __bf16* AO   = (__bf16*)alloc((size_t)M * Dm * 2);
  float*  Xr   = (float*) alloc((size_t)M * Dm * 4);
  __bf16* XN   = (__bf16*)alloc((size_t)M * Dm * 2);
  __bf16* H    = (__bf16*)alloc((size_t)M * DF * 2);

  // Weight casts.
  tb_cast_bf16<<<(Dm * Dm + 255) / 256, 256, 0, stream>>>(Wq, Wq_b, Dm * Dm);
  tb_cast_bf16<<<(Dm * Dm + 255) / 256, 256, 0, stream>>>(Wk, Wk_b, Dm * Dm);
  tb_cast_bf16<<<(Dm * Dm + 255) / 256, 256, 0, stream>>>(Wv, Wv_b, Dm * Dm);
  tb_cast_bf16<<<(Dm * Dm + 255) / 256, 256, 0, stream>>>(Wo, Wo_b, Dm * Dm);
  tb_cast_bf16<<<(Dm * DF + 255) / 256, 256, 0, stream>>>(W1, W1_b, Dm * DF);
  tb_cast_bf16<<<(DF * Dm + 255) / 256, 256, 0, stream>>>(W2, W2_b, DF * Dm);

  // Input layernorms.
  tb_layernorm<<<dim3(M), 128, 0, stream>>>(q, lnqw, lnqb, qn);
  tb_layernorm<<<dim3(M), 128, 0, stream>>>(k, lnkw, lnkb, kn);

  // Projections.
  dim3 g512(Dm / 128, M / 64);
  tb_gemm<0><<<g512, 256, 0, stream>>>(qn, Wq_b, bq, nullptr, Qp, M, Dm, Dm);
  tb_gemm<0><<<g512, 256, 0, stream>>>(kn, Wk_b, bk, nullptr, Kp, M, Dm, Dm);
  tb_gemm<0><<<g512, 256, 0, stream>>>(kn, Wv_b, bv, nullptr, Vp, M, Dm, Dm);

  // RoPE on Q and K.
  const int rope_total = B * S * 256;
  tb_rope<<<(rope_total + 255) / 256, 256, 0, stream>>>(Qp, S, rope_total);
  tb_rope<<<(rope_total + 255) / 256, 256, 0, stream>>>(Kp, S, rope_total);

  // Flash attention.
  tb_flash_attn<<<dim3(S / 64, B), 128, 0, stream>>>(Qp, Kp, Vp, AO, S);

  // Output projection + residual (x = q + out@Wo + bo), fp32.
  tb_gemm<2><<<g512, 256, 0, stream>>>(AO, Wo_b, bo, q, Xr, M, Dm, Dm);

  // Post-attention layernorm.
  tb_layernorm<<<dim3(M), 128, 0, stream>>>(Xr, lnow, lnob, XN);

  // MLP.
  dim3 gdf(DF / 128, M / 64);
  tb_gemm<1><<<gdf, 256, 0, stream>>>(XN, W1_b, b1, nullptr, H, M, DF, Dm);
  tb_gemm<2><<<g512, 256, 0, stream>>>(H, W2_b, b2, Xr, d_out, M, Dm, DF);

  (void)in_sizes; (void)n_in; (void)out_size; (void)ws_size;
}
